// Decoder_6296422056508
// MI455X (gfx1250) — compile-verified, matrix operand order checked by reading, not run on hardware
//
#include <hip/hip_runtime.h>
#include <hip/hip_bf16.h>

// ---------------------------------------------------------------------------
// Types for CDNA5 WMMA (wave32, v_wmma_f32_16x16x32_bf16)
// ---------------------------------------------------------------------------
typedef __bf16 bf16_t;
typedef __attribute__((ext_vector_type(16))) __bf16 v16bf;
typedef __attribute__((ext_vector_type(8)))  __bf16 v8bf;
typedef __attribute__((ext_vector_type(8)))  float  v8f;

#define HH   512
#define H4   2048
#define BB   256
#define UU   16
#define TT   16
#define VV   512
#define BU   4096      // B*U
#define NROW 65536     // B*U*T

__device__ __forceinline__ float sigmf(float x) { return 1.f / (1.f + __expf(-x)); }

// Load one 16x32 bf16 fragment (A-layout; B uses the same lane/K mapping with
// N playing the role of M).  Per ISA 7.12.2: lane r=lane&15 holds row r; group
// g=lane>>4 selects K offsets {g*8..g*8+7, 16+g*8..16+g*8+7} -> two 16B chunks.
__device__ __forceinline__ v16bf load_frag(const bf16_t* __restrict__ base,
                                           int ld, int row0, int k, int lane) {
    int r = lane & 15, g = lane >> 4;
    const bf16_t* p = base + (size_t)(row0 + r) * ld + k + g * 8;
    v8bf lo = *(const v8bf*)p;
    v8bf hi = *(const v8bf*)(p + 16);
    v16bf out;
#pragma unroll
    for (int i = 0; i < 8; ++i) { out[i] = lo[i]; out[i + 8] = hi[i]; }
    return out;
}

// Same fragment fetch out of an LDS-resident 64x32 bf16 panel (row-major).
__device__ __forceinline__ v16bf load_frag_lds(const bf16_t* panel, int tn, int lane) {
    int r = lane & 15, g = lane >> 4;
    const bf16_t* p = panel + (tn * 16 + r) * 32 + g * 8;
    v8bf lo = *(const v8bf*)p;
    v8bf hi = *(const v8bf*)(p + 16);
    v16bf out;
#pragma unroll
    for (int i = 0; i < 8; ++i) { out[i] = lo[i]; out[i + 8] = hi[i]; }
    return out;
}

// CDNA5 async global->LDS copy of 16 bytes (ASYNCcnt-tracked, no VGPR data).
__device__ __forceinline__ void async_copy16(unsigned lds_off, const bf16_t* g) {
    unsigned long long ga = (unsigned long long)(uintptr_t)g;
    asm volatile("global_load_async_to_lds_b128 %0, %1, off"
                 :: "v"(lds_off), "v"(ga) : "memory");
}
__device__ __forceinline__ void wait_async0() {
    asm volatile("s_wait_asynccnt 0x0" ::: "memory");
}

// ---------------------------------------------------------------------------
// Generic bf16 GEMM: Y(MxN) = act( A(MxK) @ W(NxK)^T + bias ), fp32 out.
// Block = 128 threads = 4 waves stacked on M; block tile 64(M) x 64(N).
// The 64x32 B panel (shared by all 4 waves) is double-buffered in LDS and
// filled with global_load_async_to_lds_b128; compute overlaps the prefetch.
// Requires M%64==0, N%64==0, K%32==0 (true for all uses here).
// ---------------------------------------------------------------------------
__global__ void gemm_bf16_kernel(const bf16_t* __restrict__ A, int lda,
                                 const bf16_t* __restrict__ W, int ldw,
                                 const float* __restrict__ bias,
                                 float* __restrict__ Y, long long ldy,
                                 int M, int N, int K, int act) {
    __shared__ bf16_t Bp[2][64 * 32];

    int wave = threadIdx.x >> 5;
    int lane = threadIdx.x & 31;
    int m0 = (blockIdx.x * 4 + wave) * 16;
    int n0 = blockIdx.y * 64;

    // cooperative stage of one 64(N)x32(K) panel: 4KB, 32B per thread
    int o    = threadIdx.x * 32;           // byte offset in panel
    int srow = o >> 6;                     // 64B per panel row
    int scol = (o & 63) >> 1;              // element offset in row

    v8f acc0 = {}, acc1 = {}, acc2 = {}, acc3 = {};

    {   // prologue: stage k=0
        const bf16_t* gsrc = W + (size_t)(n0 + srow) * ldw + scol;
        unsigned ldst = (unsigned)(uintptr_t)&Bp[0][o >> 1];
        async_copy16(ldst, gsrc);
        async_copy16(ldst + 16, gsrc + 8);
        wait_async0();
        __syncthreads();
    }

    int cur = 0;
    for (int k = 0; k < K; k += 32) {
        bool more = (k + 32) < K;
        if (more) {   // prefetch next panel into the other buffer
            const bf16_t* gsrc = W + (size_t)(n0 + srow) * ldw + (k + 32) + scol;
            unsigned ldst = (unsigned)(uintptr_t)&Bp[cur ^ 1][o >> 1];
            async_copy16(ldst, gsrc);
            async_copy16(ldst + 16, gsrc + 8);
        }
        v16bf a = load_frag(A, lda, m0, k, lane);
        v16bf b0 = load_frag_lds(&Bp[cur][0], 0, lane);
        v16bf b1 = load_frag_lds(&Bp[cur][0], 1, lane);
        v16bf b2 = load_frag_lds(&Bp[cur][0], 2, lane);
        v16bf b3 = load_frag_lds(&Bp[cur][0], 3, lane);
        acc0 = __builtin_amdgcn_wmma_f32_16x16x32_bf16(false, a, false, b0, (short)0, acc0, false, false);
        acc1 = __builtin_amdgcn_wmma_f32_16x16x32_bf16(false, a, false, b1, (short)0, acc1, false, false);
        acc2 = __builtin_amdgcn_wmma_f32_16x16x32_bf16(false, a, false, b2, (short)0, acc2, false, false);
        acc3 = __builtin_amdgcn_wmma_f32_16x16x32_bf16(false, a, false, b3, (short)0, acc3, false, false);
        if (more) {
            wait_async0();       // prefetch ran under the WMMAs above
            __syncthreads();
            cur ^= 1;
        }
    }

    int g = lane >> 4, c = lane & 15;
    v8f* accs[4] = { &acc0, &acc1, &acc2, &acc3 };
#pragma unroll
    for (int tn = 0; tn < 4; ++tn) {
        int col = n0 + tn * 16 + c;
        float bval = bias ? bias[col] : 0.f;
#pragma unroll
        for (int v = 0; v < 8; ++v) {
            int row = m0 + g * 8 + v;
            float y = (*accs[tn])[v] + bval;
            if (act == 1) y = tanhf(y);
            Y[(size_t)row * ldy + col] = y;
        }
    }
}

// ---------------------------------------------------------------------------
// Fused LSTM cell: g = gx(+tok table) + h@Whh^T + bhh ; c,h update.
// Block = 128 threads = 4 waves; wave w computes gate w's 16x32 tile of the
// recurrent GEMM via WMMA (K=H loop), gates staged in LDS, then elementwise.
// h in/out are bf16 (ping-pong buffers), c is fp32 in-place.
// ---------------------------------------------------------------------------
__global__ void lstm_cell_kernel(const bf16_t* __restrict__ hbf,
                                 const bf16_t* __restrict__ Whh,   // (4H,H)
                                 const float*  __restrict__ gx,    // (M,4H)
                                 const float*  __restrict__ gx_tab,// (V,4H) or null
                                 const int*    __restrict__ toks, int tok_stride,
                                 const float*  __restrict__ bhh,
                                 bf16_t* __restrict__ h_out,
                                 float*  __restrict__ c_state,
                                 bf16_t* __restrict__ seq_bf,
                                 long long seq_stride, long long seq_off,
                                 int M) {
    __shared__ float gsm[4][16][32];
    int wave = threadIdx.x >> 5;          // gate index: 0=i 1=f 2=g 3=o
    int lane = threadIdx.x & 31;
    int m0 = blockIdx.x * 16;
    int j0 = blockIdx.y * 32;
    int ncol0 = wave * HH + j0;           // column within 4H gate space

    v8f acc0 = {}, acc1 = {};
    for (int k = 0; k < HH; k += 32) {
        v16bf a  = load_frag(hbf, HH, m0, k, lane);
        v16bf b0 = load_frag(Whh, HH, ncol0, k, lane);
        v16bf b1 = load_frag(Whh, HH, ncol0 + 16, k, lane);
        acc0 = __builtin_amdgcn_wmma_f32_16x16x32_bf16(false, a, false, b0, (short)0, acc0, false, false);
        acc1 = __builtin_amdgcn_wmma_f32_16x16x32_bf16(false, a, false, b1, (short)0, acc1, false, false);
    }

    int g = lane >> 4, c = lane & 15;
#pragma unroll
    for (int v = 0; v < 8; ++v) {
        int row = m0 + g * 8 + v;
        int tok = 0;
        if (gx_tab) tok = toks[(size_t)row * tok_stride];
#pragma unroll
        for (int h = 0; h < 2; ++h) {
            int col = ncol0 + h * 16 + c;
            float val = (h ? acc1[v] : acc0[v]) + gx[(size_t)row * H4 + col] + bhh[col];
            if (gx_tab) val += gx_tab[(size_t)tok * H4 + col];
            gsm[wave][g * 8 + v][h * 16 + c] = val;
        }
    }
    __syncthreads();

    for (int e = threadIdx.x; e < 512; e += 128) {
        int r = e >> 5, cc = e & 31;
        float iv = gsm[0][r][cc];
        float fv = gsm[1][r][cc];
        float gv = gsm[2][r][cc];
        float ov = gsm[3][r][cc];
        size_t idx = (size_t)(m0 + r) * HH + (j0 + cc);
        float cold = c_state[idx];
        float cnew = sigmf(fv) * cold + sigmf(iv) * tanhf(gv);
        float hnew = sigmf(ov) * tanhf(cnew);
        c_state[idx] = cnew;
        h_out[idx]   = (bf16_t)hnew;
        if (seq_bf)
            seq_bf[(size_t)(m0 + r) * seq_stride + seq_off + (j0 + cc)] = (bf16_t)hnew;
    }
}

// ---------------------------------------------------------------------------
// Elementwise helpers
// ---------------------------------------------------------------------------
__global__ void cvt_bf16_kernel(const float* __restrict__ in,
                                bf16_t* __restrict__ out, long long n) {
    long long i = (long long)blockIdx.x * blockDim.x + threadIdx.x;
    if (i < n) out[i] = (bf16_t)in[i];
}

__global__ void zero_bf_kernel(bf16_t* __restrict__ p, long long n) {
    long long i = (long long)blockIdx.x * blockDim.x + threadIdx.x;
    if (i < n) p[i] = (bf16_t)0.f;
}

// ci(B,4H) -> h1,h2 (bf16 states) + c1,c2 (fp32 states)
__global__ void split_ci_kernel(const float* __restrict__ ci,
                                bf16_t* __restrict__ h1, bf16_t* __restrict__ h2,
                                float* __restrict__ c1, float* __restrict__ c2,
                                long long n) {
    long long i = (long long)blockIdx.x * blockDim.x + threadIdx.x;
    if (i >= n) return;
    long long b = i / HH, j = i % HH;
    const float* r = ci + b * H4;
    h1[i] = (bf16_t)r[j];
    h2[i] = (bf16_t)r[HH + j];
    c1[i] = r[2 * HH + j];
    c2[i] = r[3 * HH + j];
}

// states(BU,4H) -> dh1,dh2 (bf16), dc1,dc2 (fp32), hinit = dh1+dh2 (bf16)
__global__ void split_states_kernel(const float* __restrict__ st,
                                    bf16_t* __restrict__ h0, bf16_t* __restrict__ h1,
                                    float* __restrict__ c0, float* __restrict__ c1,
                                    bf16_t* __restrict__ hinit, long long n) {
    long long i = (long long)blockIdx.x * blockDim.x + threadIdx.x;
    if (i >= n) return;
    long long b = i / HH, j = i % HH;
    const float* r = st + b * H4;
    float a = r[j], d = r[HH + j];
    h0[i] = (bf16_t)a;
    h1[i] = (bf16_t)d;
    c0[i] = r[2 * HH + j];
    c1[i] = r[3 * HH + j];
    hinit[i] = (bf16_t)(a + d);
}

// Row softmax, one block (256 thr) per row of length VV
__global__ void softmax_kernel(float* __restrict__ data) {
    __shared__ float red[256];
    float* p = data + (size_t)blockIdx.x * VV;
    float mx = -1e30f;
    for (int j = threadIdx.x; j < VV; j += 256) mx = fmaxf(mx, p[j]);
    red[threadIdx.x] = mx; __syncthreads();
    for (int s = 128; s > 0; s >>= 1) {
        if (threadIdx.x < s) red[threadIdx.x] = fmaxf(red[threadIdx.x], red[threadIdx.x + s]);
        __syncthreads();
    }
    mx = red[0]; __syncthreads();
    float sum = 0.f;
    for (int j = threadIdx.x; j < VV; j += 256) {
        float e = __expf(p[j] - mx); p[j] = e; sum += e;
    }
    red[threadIdx.x] = sum; __syncthreads();
    for (int s = 128; s > 0; s >>= 1) {
        if (threadIdx.x < s) red[threadIdx.x] += red[threadIdx.x + s];
        __syncthreads();
    }
    float inv = 1.f / red[0];
    for (int j = threadIdx.x; j < VV; j += 256) p[j] *= inv;
}

// ---------------------------------------------------------------------------
// Host orchestration
// ---------------------------------------------------------------------------
extern "C" void kernel_launch(void* const* d_in, const int* in_sizes, int n_in,
                              void* d_out, int out_size, void* d_ws, size_t ws_size,
                              hipStream_t stream) {
    (void)in_sizes; (void)n_in; (void)out_size; (void)ws_size;

    const float* z     = (const float*)d_in[0];
    const int*   x     = (const int*)  d_in[1];
    const float* Wc    = (const float*)d_in[2];
    const float* bc    = (const float*)d_in[3];
    const float* cWih0 = (const float*)d_in[4];
    const float* cWhh0 = (const float*)d_in[5];
    const float* cbih0 = (const float*)d_in[6];
    const float* cbhh0 = (const float*)d_in[7];
    const float* cWih1 = (const float*)d_in[8];
    const float* cWhh1 = (const float*)d_in[9];
    const float* cbih1 = (const float*)d_in[10];
    const float* cbhh1 = (const float*)d_in[11];
    const float* Wd    = (const float*)d_in[12];
    const float* bd    = (const float*)d_in[13];
    const float* dWih0 = (const float*)d_in[14];
    const float* dWhh0 = (const float*)d_in[15];
    const float* dbih0 = (const float*)d_in[16];
    const float* dbhh0 = (const float*)d_in[17];
    const float* dWih1 = (const float*)d_in[18];
    const float* dWhh1 = (const float*)d_in[19];
    const float* dbih1 = (const float*)d_in[20];
    const float* dbhh1 = (const float*)d_in[21];
    const float* emb   = (const float*)d_in[22];
    const float* Wo    = (const float*)d_in[23];
    const float* bo    = (const float*)d_in[24];
    float* out = (float*)d_out;

    char* wsb = (char*)d_ws;
    size_t off = 0;
    auto alloc = [&](size_t bytes) -> void* {
        void* p = wsb + off;
        off += (bytes + 255) & ~(size_t)255;
        return p;
    };
    auto cvt = [&](const float* src, size_t n) -> bf16_t* {
        bf16_t* dst = (bf16_t*)alloc(n * sizeof(bf16_t));
        cvt_bf16_kernel<<<dim3((unsigned)((n + 255) / 256)), dim3(256), 0, stream>>>(src, dst, (long long)n);
        return dst;
    };
    auto gemm = [&](const bf16_t* A, int lda, const bf16_t* W, int ldw,
                    const float* bias, float* Y, long long ldy,
                    int M, int N, int K, int act) {
        gemm_bf16_kernel<<<dim3(M / 64, N / 64), dim3(128), 0, stream>>>(
            A, lda, W, ldw, bias, Y, ldy, M, N, K, act);
    };
    auto cell = [&](const bf16_t* hbf, const bf16_t* Whh, const float* gx,
                    const float* tab, const int* toks, int tok_stride,
                    const float* bhh, bf16_t* h_out, float* c_state,
                    bf16_t* seq, long long seq_stride, long long seq_off, int M) {
        lstm_cell_kernel<<<dim3(M / 16, HH / 32), dim3(128), 0, stream>>>(
            hbf, Whh, gx, tab, toks, tok_stride, bhh, h_out, c_state,
            seq, seq_stride, seq_off, M);
    };

    // ---- bf16 weight/input copies (resident in 192MB L2 afterwards) ----
    bf16_t* z_b     = cvt(z,     (size_t)BB * HH);
    bf16_t* Wc_b    = cvt(Wc,    (size_t)H4 * HH);
    bf16_t* cWih0_b = cvt(cWih0, (size_t)H4 * HH);
    bf16_t* cWhh0_b = cvt(cWhh0, (size_t)H4 * HH);
    bf16_t* cWih1_b = cvt(cWih1, (size_t)H4 * HH);
    bf16_t* cWhh1_b = cvt(cWhh1, (size_t)H4 * HH);
    bf16_t* Wd_b    = cvt(Wd,    (size_t)H4 * HH);
    bf16_t* dWih0_b = cvt(dWih0, (size_t)H4 * 2 * HH);
    bf16_t* dWhh0_b = cvt(dWhh0, (size_t)H4 * HH);
    bf16_t* dWih1_b = cvt(dWih1, (size_t)H4 * HH);
    bf16_t* dWhh1_b = cvt(dWhh1, (size_t)H4 * HH);
    bf16_t* Wo_b    = cvt(Wo,    (size_t)VV * HH);
    bf16_t* emb_b   = cvt(emb,   (size_t)VV * HH);

    // ---- scratch ----
    float*  ci    = (float*)alloc((size_t)BB * H4 * 4);
    float*  gxc   = (float*)alloc((size_t)BB * H4 * 4);
    bf16_t* ch1[2], *ch2[2];
    ch1[0] = (bf16_t*)alloc((size_t)BB * HH * 2);
    ch1[1] = (bf16_t*)alloc((size_t)BB * HH * 2);
    ch2[0] = (bf16_t*)alloc((size_t)BB * HH * 2);
    ch2[1] = (bf16_t*)alloc((size_t)BB * HH * 2);
    float*  cc1   = (float*)alloc((size_t)BB * HH * 4);
    float*  cc2   = (float*)alloc((size_t)BB * HH * 4);
    bf16_t* o0b   = (bf16_t*)alloc((size_t)BB * HH * 2);
    bf16_t* cond_out = (bf16_t*)alloc((size_t)BU * HH * 2);
    float*  states   = (float*)alloc((size_t)BU * H4 * 4);
    bf16_t* dl0h[2], *dl1h[2];
    dl0h[0] = (bf16_t*)alloc((size_t)BU * HH * 2);
    dl0h[1] = (bf16_t*)alloc((size_t)BU * HH * 2);
    dl1h[0] = (bf16_t*)alloc((size_t)BU * HH * 2);
    dl1h[1] = (bf16_t*)alloc((size_t)BU * HH * 2);
    float*  dl0c  = (float*)alloc((size_t)BU * HH * 4);
    float*  dl1c  = (float*)alloc((size_t)BU * HH * 4);
    bf16_t* hinit = (bf16_t*)alloc((size_t)BU * HH * 2);
    float*  Temb  = (float*)alloc((size_t)VV * H4 * 4);
    float*  Hpart = (float*)alloc((size_t)BU * H4 * 4);
    float*  gx1   = (float*)alloc((size_t)BU * H4 * 4);

    zero_bf_kernel<<<dim3((BB * HH + 255) / 256), dim3(256), 0, stream>>>(o0b, (long long)BB * HH);

    // ---- ci = tanh(z @ Wc^T + bc); split into cond initial states ----
    gemm(z_b, HH, Wc_b, HH, bc, ci, H4, BB, H4, HH, 1);
    split_ci_kernel<<<dim3((BB * HH + 255) / 256), dim3(256), 0, stream>>>(
        ci, ch1[0], ch2[0], cc1, cc2, (long long)BB * HH);

    // ---- conditioning LSTM: U steps of 2 cells, seq out -> cond_out ----
    int p1 = 0, p2 = 0;
    for (int u = 0; u < UU; ++u) {
        const bf16_t* o_in = (u == 0) ? o0b : ch2[p2];
        gemm(o_in, HH, cWih0_b, HH, cbih0, gxc, H4, BB, H4, HH, 0);
        cell(ch1[p1], cWhh0_b, gxc, nullptr, nullptr, 0, cbhh0,
             ch1[p1 ^ 1], cc1, nullptr, 0, 0, BB);
        p1 ^= 1;
        gemm(ch1[p1], HH, cWih1_b, HH, cbih1, gxc, H4, BB, H4, HH, 0);
        cell(ch2[p2], cWhh1_b, gxc, nullptr, nullptr, 0, cbhh1,
             ch2[p2 ^ 1], cc2, cond_out, (long long)UU * HH, (long long)u * HH, BB);
        p2 ^= 1;
    }

    // ---- states = tanh(cond_out @ Wd^T + bd); split decoder states ----
    gemm(cond_out, HH, Wd_b, HH, bd, states, H4, BU, H4, HH, 1);
    split_states_kernel<<<dim3((BU * HH + 255) / 256), dim3(256), 0, stream>>>(
        states, dl0h[0], dl1h[0], dl0c, dl1c, hinit, (long long)BU * HH);

    // ---- precomputed layer-0 x-gates: table + time-invariant part ----
    // Temb  = emb @ dWih0[:, :H]^T               (token lookup table)
    gemm(emb_b, HH, dWih0_b, 2 * HH, nullptr, Temb, H4, VV, H4, HH, 0);
    // Hpart = hinit @ dWih0[:, H:]^T + dbih0     (same for all t)
    gemm(hinit, HH, dWih0_b + HH, 2 * HH, dbih0, Hpart, H4, BU, H4, HH, 0);

    // ---- decoder LSTM, layers interleaved per timestep ----
    int q0 = 0, q1 = 0;
    for (int t = 0; t < TT; ++t) {
        const float* tab  = (t == 0) ? nullptr : Temb;
        const int*   toks = (t == 0) ? nullptr : (x + (t - 1));
        cell(dl0h[q0], dWhh0_b, Hpart, tab, toks, TT, dbhh0,
             dl0h[q0 ^ 1], dl0c, nullptr, 0, 0, BU);
        q0 ^= 1;
        gemm(dl0h[q0], HH, dWih1_b, HH, dbih1, gx1, H4, BU, H4, HH, 0);
        cell(dl1h[q1], dWhh1_b, gx1, nullptr, nullptr, 0, dbhh1,
             dl1h[q1 ^ 1], dl1c, nullptr, 0, 0, BU);
        q1 ^= 1;
        // logits for this t straight into d_out: row n -> out[(n*T + t)*V]
        gemm(dl1h[q1], HH, Wo_b, HH, bo, out + (size_t)t * VV,
             (long long)TT * VV, BU, VV, HH, 0);
    }

    // ---- softmax over all B*U*T rows ----
    softmax_kernel<<<dim3(NROW), dim3(256), 0, stream>>>(out);
}
